// DeformableConv2d_45792941310057
// MI455X (gfx1250) — matrix operand order repaired and checked by reading, not compiled, not run on hardware
//
#include <hip/hip_runtime.h>
#include <hip/hip_fp16.h>
#include <math.h>
#include <stdint.h>

typedef __attribute__((ext_vector_type(16))) _Float16 v16h;
typedef __attribute__((ext_vector_type(8)))  _Float16 v8h;
typedef __attribute__((ext_vector_type(8)))  float    v8f;
typedef __attribute__((ext_vector_type(2)))  int      v2i;

#define B_  8
#define C_  256
#define H_  64
#define W_  64
#define O_  256
#define K2_ 9
#define HW_ 4096
#define KC_ 2304   // K2_ * C_
#define GN_ 32     // pixels per GEMM workgroup
#define NCHUNK_ 72 // KC_ / 32

static __device__ __forceinline__ v16h cat8(v8h lo, v8h hi) {
  return __builtin_shufflevector(lo, hi, 0,1,2,3,4,5,6,7,8,9,10,11,12,13,14,15);
}
static __device__ __forceinline__ v8h zero8() {
  v8h z = { (_Float16)0, (_Float16)0, (_Float16)0, (_Float16)0,
            (_Float16)0, (_Float16)0, (_Float16)0, (_Float16)0 };
  return z;
}
static __device__ __forceinline__ int iclamp(int v, int lo, int hi) {
  return v < lo ? lo : (v > hi ? hi : v);
}

// ---- CDNA5 async copy to LDS (ASYNCcnt-tracked), builtin if declared else asm
static __device__ __forceinline__ void async_copy_b64(const _Float16* gsrc,
                                                      _Float16* ldst) {
#if __has_builtin(__builtin_amdgcn_global_load_async_to_lds_b64)
  __builtin_amdgcn_global_load_async_to_lds_b64(
      (__attribute__((address_space(1))) v2i*)(uintptr_t)gsrc,
      (__attribute__((address_space(3))) v2i*)(uint32_t)(uintptr_t)ldst, 0, 0);
#else
  unsigned laddr = (unsigned)(uintptr_t)ldst;             // LDS byte offset
  unsigned long long gaddr = (unsigned long long)(uintptr_t)gsrc;
  asm volatile("global_load_async_to_lds_b64 %0, %1, off"
               :: "v"(laddr), "v"(gaddr) : "memory");
#endif
}
static __device__ __forceinline__ void wait_async0() {
#if __has_builtin(__builtin_amdgcn_s_wait_asynccnt)
  __builtin_amdgcn_s_wait_asynccnt(0);
#else
  asm volatile("s_wait_asynccnt 0x0" ::: "memory");
#endif
}

// ---------------------------------------------------------------------------
// Kernel 0: NCHW fp32 -> NHWC f16 pack for x and offset_in.
// ---------------------------------------------------------------------------
__global__ __launch_bounds__(256) void pack_nhwc(
    const float* __restrict__ x, const float* __restrict__ offin,
    _Float16* __restrict__ xT_x, _Float16* __restrict__ xT_off) {
  int wave = (int)((blockIdx.x * blockDim.x + threadIdx.x) >> 5);
  int lane = (int)(threadIdx.x & 31);
  int t   = wave / (B_ * HW_);
  int pix = wave % (B_ * HW_);
  const float* src = t ? offin : x;
  _Float16* dst    = t ? xT_off : xT_x;
  int b = pix >> 12, hw = pix & (HW_ - 1);
  int c0 = lane * 8;
  v8h o;
  #pragma unroll
  for (int j = 0; j < 8; ++j)
    o[j] = (_Float16)src[((size_t)(b * C_ + c0 + j)) * HW_ + hw];
  *(v8h*)(dst + (size_t)pix * C_ + c0) = o;
}

// ---------------------------------------------------------------------------
// Kernel 1: weight repack.
//   Wr[o][k2*C + c]  = reg_w[o][c][k2]                  (256 x 2304 f16)
//   Wom[m][k2*C + c] : rows 0..17 = offset_w, rows 32..40 = mod_w, rest 0
// ---------------------------------------------------------------------------
__global__ __launch_bounds__(256) void pack_weights(
    const float* __restrict__ offw, const float* __restrict__ offb,
    const float* __restrict__ modw, const float* __restrict__ modb,
    const float* __restrict__ regw,
    _Float16* __restrict__ Wom, float* __restrict__ bias_om,
    _Float16* __restrict__ Wr) {
  int tid = (int)(blockIdx.x * blockDim.x + threadIdx.x);   // < 256*2304
  {
    int o = tid / KC_, k = tid % KC_;
    int k2 = k >> 8, c = k & 255;
    Wr[tid] = (_Float16)regw[((size_t)(o * C_ + c)) * K2_ + k2];
  }
  if (tid < 48 * KC_) {
    int m = tid / KC_, k = tid % KC_;
    int k2 = k >> 8, c = k & 255;
    float v = 0.f;
    if (m < 18)                 v = offw[((size_t)(m * C_ + c)) * K2_ + k2];
    else if (m >= 32 && m < 41) v = modw[((size_t)((m - 32) * C_ + c)) * K2_ + k2];
    Wom[tid] = (_Float16)v;
  }
  if (tid < 48) {
    float v = 0.f;
    if (tid < 18)                   v = offb[tid];
    else if (tid >= 32 && tid < 41) v = modb[tid - 32];
    bias_om[tid] = v;
  }
}

// ---------------------------------------------------------------------------
// Kernel 2: offset + modulator conv via WMMA implicit GEMM.
// ---------------------------------------------------------------------------
__global__ __launch_bounds__(256) void conv_offmod(
    const _Float16* __restrict__ xT_x, const _Float16* __restrict__ xT_off,
    const _Float16* __restrict__ Wom, const float* __restrict__ bias_om,
    float* __restrict__ offbuf, float* __restrict__ maskbuf) {
  int wave = (int)((blockIdx.x * blockDim.x + threadIdx.x) >> 5);
  int lane = (int)(threadIdx.x & 31);
  int mt = wave % 3;
  int pb = (wave / 3) & 255;
  int b  = wave / (3 * 256);
  const _Float16* inp = (mt < 2) ? xT_off : xT_x;
  int g = lane >> 4;
  int n = lane & 15;
  int p = pb * 16 + n;
  int y = p >> 6, x = p & 63;
  int mrow = mt * 16 + n;
  v8f acc = {0.f,0.f,0.f,0.f,0.f,0.f,0.f,0.f};

  for (int k2 = 0; k2 < K2_; ++k2) {
    int ky = k2 / 3, kx = k2 % 3;
    int yy = y + ky - 1, xx = x + kx - 1;
    bool ok = (yy >= 0) && (yy < H_) && (xx >= 0) && (xx < W_);
    const _Float16* bsrc = inp + ((size_t)(b * HW_ + yy * W_ + xx)) * C_;
    const _Float16* arow = Wom + ((size_t)mrow * K2_ + k2) * C_;
    #pragma unroll
    for (int cc = 0; cc < C_; cc += 32) {
      v8h alo = *(const v8h*)(arow + cc + g * 8);
      v8h ahi = *(const v8h*)(arow + cc + 16 + g * 8);
      v16h A = cat8(alo, ahi);
      v8h b0 = zero8(), b1 = zero8();
      if (ok) {
        b0 = *(const v8h*)(bsrc + cc + g * 8);
        b1 = *(const v8h*)(bsrc + cc + 16 + g * 8);
      }
      v16h Bm = cat8(b0, b1);
      acc = __builtin_amdgcn_wmma_f32_16x16x32_f16(
          false, A, false, Bm, (short)0, acc, false, false);
    }
  }

  #pragma unroll
  for (int r = 0; r < 8; ++r) {
    int ml = r + 8 * g;
    float v = acc[r] + bias_om[mt * 16 + ml];
    if (mt < 2) {
      int ch = mt * 16 + ml;
      if (ch < 18) offbuf[((size_t)(b * 18 + ch)) * HW_ + p] = v;
    } else if (ml < 9) {
      maskbuf[((size_t)(b * 9 + ml)) * HW_ + p] = 2.f / (1.f + expf(-v));
    }
  }
}

// ---------------------------------------------------------------------------
// Kernel 3: deformable im2col (bilinear + modulator) for one batch.
// ---------------------------------------------------------------------------
__global__ __launch_bounds__(256) void im2col_deform(
    int b, const _Float16* __restrict__ xT_x,
    const float* __restrict__ offbuf, const float* __restrict__ maskbuf,
    _Float16* __restrict__ col) {
  int wave = (int)((blockIdx.x * blockDim.x + threadIdx.x) >> 5);
  int lane = (int)(threadIdx.x & 31);
  int k2 = wave % 9;
  int p  = wave / 9;
  int y = p >> 6, x = p & 63;
  int ky = k2 / 3, kx = k2 % 3;

  float oy = offbuf[((size_t)(b * 18 + 2 * k2    )) * HW_ + p];
  float ox = offbuf[((size_t)(b * 18 + 2 * k2 + 1)) * HW_ + p];
  float mk = maskbuf[((size_t)(b * 9 + k2)) * HW_ + p];

  float py = oy + (float)(y + ky - 1);
  float px = ox + (float)(x + kx - 1);
  float y0f = floorf(py), x0f = floorf(px);
  float wy = py - y0f,  wx = px - x0f;
  int y0 = (int)y0f, x0 = (int)x0f;
  int y1 = y0 + 1,   x1 = x0 + 1;
  bool y0ok = (y0 >= 0) && (y0 < H_);
  bool y1ok = (y1 >= 0) && (y1 < H_);
  bool x0ok = (x0 >= 0) && (x0 < W_);
  bool x1ok = (x1 >= 0) && (x1 < W_);
  float w00 = (1.f - wy) * (1.f - wx) * mk * (float)(y0ok && x0ok);
  float w01 = (1.f - wy) * wx         * mk * (float)(y0ok && x1ok);
  float w10 = wy         * (1.f - wx) * mk * (float)(y1ok && x0ok);
  float w11 = wy         * wx         * mk * (float)(y1ok && x1ok);
  int y0c = iclamp(y0, 0, H_ - 1), y1c = iclamp(y1, 0, H_ - 1);
  int x0c = iclamp(x0, 0, W_ - 1), x1c = iclamp(x1, 0, W_ - 1);

  const _Float16* base = xT_x + (size_t)b * HW_ * C_;
  int c0 = lane * 8;
  v8h h00 = *(const v8h*)(base + ((size_t)(y0c * W_ + x0c)) * C_ + c0);
  v8h h01 = *(const v8h*)(base + ((size_t)(y0c * W_ + x1c)) * C_ + c0);
  v8h h10 = *(const v8h*)(base + ((size_t)(y1c * W_ + x0c)) * C_ + c0);
  v8h h11 = *(const v8h*)(base + ((size_t)(y1c * W_ + x1c)) * C_ + c0);
  v8h outv;
  #pragma unroll
  for (int j = 0; j < 8; ++j) {
    float v = w00 * (float)h00[j] + w01 * (float)h01[j]
            + w10 * (float)h10[j] + w11 * (float)h11[j];
    outv[j] = (_Float16)v;
  }
  *(v8h*)(col + ((size_t)p * K2_ + k2) * C_ + c0) = outv;
}

// ---------------------------------------------------------------------------
// Kernel 4: main GEMM for one batch, LDS-staged B with async double-buffering.
// Workgroup (8 waves) computes M=256 (all out-ch) x N=32 pixels, so each col
// element is fetched from global exactly once per batch. Per K-chunk of 32,
// the 2KB B panel [32 px][32 k] is async-copied to LDS (b64 per thread,
// ASYNCcnt), double buffered; fragments are read back with ds_load_b128.
// Wave w owns M-tiles {2w, 2w+1} x N-tiles {0,1}: 4 WMMAs per chunk.
// ---------------------------------------------------------------------------
__global__ __launch_bounds__(256) void gemm_deform(
    int b, const _Float16* __restrict__ col, const _Float16* __restrict__ Wr,
    float* __restrict__ out) {
  __shared__ _Float16 lbuf[2][GN_ * 32];     // 2 x 2KB

  int tid  = (int)threadIdx.x;
  int wave = tid >> 5, lane = tid & 31;
  int nb = (int)blockIdx.x;                  // 0..127, 32 pixels each
  int g = lane >> 4, n = lane & 15;
  int mt0 = wave * 2;

  const _Float16* colb = col + (size_t)(nb * GN_) * KC_;

  // copy assignment: thread copies 8B (4 halves): px = tid>>3, q = tid&7
  int cp_px = tid >> 3, cp_q = tid & 7;
  const _Float16* cp_g = colb + (size_t)cp_px * KC_ + cp_q * 4;
  int cp_l = cp_px * 32 + cp_q * 4;          // halves within a buffer

  const _Float16* arow0 = Wr + (size_t)(mt0 * 16 + n) * KC_;
  const _Float16* arow1 = Wr + (size_t)((mt0 + 1) * 16 + n) * KC_;

  v8f acc[4];
  #pragma unroll
  for (int s = 0; s < 4; ++s) {
    v8f z = {0.f,0.f,0.f,0.f,0.f,0.f,0.f,0.f};
    acc[s] = z;
  }

  // prologue: stage chunk 0 into buffer 0
  async_copy_b64(cp_g, &lbuf[0][cp_l]);

  for (int kc = 0; kc < NCHUNK_; ++kc) {
    wait_async0();          // own async copies (chunk kc) landed
    __syncthreads();        // everyone's copies landed / last reads done
    if (kc + 1 < NCHUNK_)
      async_copy_b64(cp_g + (kc + 1) * 32, &lbuf[(kc + 1) & 1][cp_l]);

    const _Float16* lb = &lbuf[kc & 1][0];
    v16h A0 = cat8(*(const v8h*)(arow0 + kc * 32 + g * 8),
                   *(const v8h*)(arow0 + kc * 32 + 16 + g * 8));
    v16h A1 = cat8(*(const v8h*)(arow1 + kc * 32 + g * 8),
                   *(const v8h*)(arow1 + kc * 32 + 16 + g * 8));
    v16h B0 = cat8(*(const v8h*)(lb + n * 32 + g * 8),
                   *(const v8h*)(lb + n * 32 + 16 + g * 8));
    v16h B1 = cat8(*(const v8h*)(lb + (16 + n) * 32 + g * 8),
                   *(const v8h*)(lb + (16 + n) * 32 + 16 + g * 8));
    acc[0] = __builtin_amdgcn_wmma_f32_16x16x32_f16(false, A0, false, B0,
                                                    (short)0, acc[0], false, false);
    acc[1] = __builtin_amdgcn_wmma_f32_16x16x32_f16(false, A0, false, B1,
                                                    (short)0, acc[1], false, false);
    acc[2] = __builtin_amdgcn_wmma_f32_16x16x32_f16(false, A1, false, B0,
                                                    (short)0, acc[2], false, false);
    acc[3] = __builtin_amdgcn_wmma_f32_16x16x32_f16(false, A1, false, B1,
                                                    (short)0, acc[3], false, false);
  }

  float* outb = out + (size_t)b * O_ * HW_;
  #pragma unroll
  for (int mi = 0; mi < 2; ++mi) {
    #pragma unroll
    for (int ni = 0; ni < 2; ++ni) {
      int p = nb * GN_ + ni * 16 + n;
      #pragma unroll
      for (int r = 0; r < 8; ++r) {
        int o = (mt0 + mi) * 16 + r + 8 * g;
        outb[(size_t)o * HW_ + p] = acc[mi * 2 + ni][r];
      }
    }
  }
}

// ---------------------------------------------------------------------------
extern "C" void kernel_launch(void* const* d_in, const int* in_sizes, int n_in,
                              void* d_out, int out_size, void* d_ws, size_t ws_size,
                              hipStream_t stream) {
  const float* x     = (const float*)d_in[0];
  const float* offin = (const float*)d_in[1];
  const float* offw  = (const float*)d_in[2];
  const float* offb  = (const float*)d_in[3];
  const float* modw  = (const float*)d_in[4];
  const float* modb  = (const float*)d_in[5];
  const float* regw  = (const float*)d_in[6];
  float* out = (float*)d_out;

  char* w = (char*)d_ws;
  _Float16* xT_x    = (_Float16*)w; w += (size_t)B_ * HW_ * C_ * 2;   // 16.8 MB
  _Float16* xT_off  = (_Float16*)w; w += (size_t)B_ * HW_ * C_ * 2;   // 16.8 MB
  _Float16* Wom     = (_Float16*)w; w += (size_t)48 * KC_ * 2;        // 216 KB
  float*    bias_om = (float*)w;    w += 256;
  _Float16* Wr      = (_Float16*)w; w += (size_t)O_ * KC_ * 2;        // 1.2 MB
  float*    offbuf  = (float*)w;    w += (size_t)B_ * 18 * HW_ * 4;   // 2.4 MB
  float*    maskbuf = (float*)w;    w += (size_t)B_ * 9 * HW_ * 4;    // 1.2 MB
  _Float16* col     = (_Float16*)w;                                   // 18.9 MB

  pack_nhwc<<<(2 * B_ * HW_) / 8, 256, 0, stream>>>(x, offin, xT_x, xT_off);
  pack_weights<<<(O_ * KC_) / 256, 256, 0, stream>>>(offw, offb, modw, modb,
                                                     regw, Wom, bias_om, Wr);
  conv_offmod<<<(B_ * 256 * 3) / 8, 256, 0, stream>>>(xT_x, xT_off, Wom,
                                                      bias_om, offbuf, maskbuf);
  for (int b = 0; b < B_; ++b) {
    im2col_deform<<<(HW_ * K2_) / 8, 256, 0, stream>>>(b, xT_x, offbuf, maskbuf, col);
    gemm_deform<<<HW_ / GN_, 256, 0, stream>>>(b, col, Wr, out);
  }
}
